// KeywordsAttention_87909390614752
// MI455X (gfx1250) — matrix operand order repaired and checked by reading, not compiled, not run on hardware
//
#include <hip/hip_runtime.h>
#include <math.h>

#define BATCH 64
#define SRC   2048
#define DIM   1024

typedef __attribute__((ext_vector_type(16))) __bf16 v16bf;
typedef __attribute__((ext_vector_type(8)))  __bf16 v8bf;
typedef __attribute__((ext_vector_type(8)))  float  v8f;

// padded LDS strides (bf16 elems) to avoid bank conflicts on fragment reads
#define ASTRIDE 1032   // +4 dwords per row -> 4-bank rotation (16B reads)
#define BSTRIDE 1040   // +8 dwords per row -> 8-bank rotation (32B reads)

#if __has_builtin(__builtin_amdgcn_tanhf)
#define FAST_TANH(x) __builtin_amdgcn_tanhf(x)   // v_tanh_f32 (TRANS unit)
#else
#define FAST_TANH(x) tanhf(x)
#endif

// ---------------------------------------------------------------------------
// K0: bias[b,e] = b_query[e] + source[b,:]·W_query[e,:] + condition[b,:]·W_cond[e,:]
// ---------------------------------------------------------------------------
__global__ __launch_bounds__(256)
void bias_kernel(const float* __restrict__ src, const float* __restrict__ cond,
                 const float* __restrict__ Wq, const float* __restrict__ bq,
                 const float* __restrict__ Wc, float* __restrict__ bias) {
    int idx = blockIdx.x * 256 + threadIdx.x;      // 0 .. 65535
    int b = idx >> 10;
    int e = idx & (DIM - 1);
    const float* s  = src  + (size_t)b * DIM;
    const float* c  = cond + (size_t)b * DIM;
    const float* wq = Wq   + (size_t)e * DIM;
    const float* wc = Wc   + (size_t)e * DIM;
    float ax = 0.f, ay = 0.f, az = 0.f, aw = 0.f;
    for (int d = 0; d < DIM; d += 4) {
        float4 sv = *(const float4*)(s + d);
        float4 cv = *(const float4*)(c + d);
        float4 qv = *(const float4*)(wq + d);
        float4 wv = *(const float4*)(wc + d);
        ax += sv.x * qv.x + cv.x * wv.x;
        ay += sv.y * qv.y + cv.y * wv.y;
        az += sv.z * qv.z + cv.z * wv.z;
        aw += sv.w * qv.w + cv.w * wv.w;
    }
    bias[idx] = bq[e] + ((ax + ay) + (az + aw));
}

// ---------------------------------------------------------------------------
// K1: scores[b,s] = sum_e v[e] * tanh(bias[b,e] + sum_d mb[b,s,d]*Wctx[e,d])
//     Block = (batch b, 128-row s tile), 8 waves, bf16 WMMA 16x16x32.
//     A strip resident full-K in LDS (read HBM once); B tiles double-buffered
//     through registers so L2 traffic overlaps WMMA compute.
// ---------------------------------------------------------------------------
__global__ __launch_bounds__(256)
void score_kernel(const float* __restrict__ mb, const float* __restrict__ Wctx,
                  const float* __restrict__ bias, const float* __restrict__ v,
                  float* __restrict__ scores) {
    __shared__ __bf16 Alds[128 * ASTRIDE];   // 258 KB: full-K A strip (bf16)
    __shared__ __bf16 Blds[16  * BSTRIDE];   // 33 KB: current W_ctx 16-e tile

    const int b    = blockIdx.x;
    const int st   = blockIdx.y;             // 16 s-tiles of 128
    const int tid  = threadIdx.x;
    const int wave = tid >> 5;               // 8 waves -> 16 s rows each
    const int lane = tid & 31;
    const int m    = lane & 15;              // M (A) / N (B,C) index
    const int h    = lane >> 4;              // half-wave selector

    // ---- cooperative load: mb[b, st*128 : +128, :] -> bf16 LDS (HBM read once)
    const float* Ag = mb + ((size_t)b * SRC + (size_t)st * 128) * DIM;
    for (int i = tid; i < 128 * (DIM / 4); i += 256) {
        int row = i >> 8;                    // 256 float4 per row
        int c4  = (i & 255) << 2;
        float4 f = *(const float4*)(Ag + (size_t)row * DIM + c4);
        __bf16* dst = &Alds[row * ASTRIDE + c4];
        dst[0] = (__bf16)f.x; dst[1] = (__bf16)f.y;
        dst[2] = (__bf16)f.z; dst[3] = (__bf16)f.w;
    }

    // ---- register-staged B prefetch: thread tid holds row j, cols tid*4..+3
    const int bcol = tid << 2;               // float col within a W_ctx row
    float4 stage[16];
#pragma unroll
    for (int j = 0; j < 16; ++j)
        stage[j] = *(const float4*)(Wctx + (size_t)j * DIM + bcol);

    float sAcc[8];
#pragma unroll
    for (int r = 0; r < 8; ++r) sAcc[r] = 0.f;

    const __bf16* arow = &Alds[(wave * 16 + m) * ASTRIDE];
    const __bf16* brow = &Blds[m * BSTRIDE];
    const float*  biasb = bias + (size_t)b * DIM;

    for (int et = 0; et < DIM / 16; ++et) {
        __syncthreads();                     // all waves done reading Blds
        // commit staged tile et to LDS (fp32 -> bf16 at store time)
#pragma unroll
        for (int j = 0; j < 16; ++j) {
            __bf16* dst = &Blds[j * BSTRIDE + bcol];
            dst[0] = (__bf16)stage[j].x; dst[1] = (__bf16)stage[j].y;
            dst[2] = (__bf16)stage[j].z; dst[3] = (__bf16)stage[j].w;
        }
        // issue prefetch of tile et+1; lands while we do the WMMAs below
        if (et + 1 < DIM / 16) {
            const float* Bg = Wctx + (size_t)(et + 1) * 16 * DIM;
#pragma unroll
            for (int j = 0; j < 16; ++j)
                stage[j] = *(const float4*)(Bg + (size_t)j * DIM + bcol);
        }
        __syncthreads();                     // Blds tile et visible

        v8f acc = {};
#pragma unroll 4
        for (int kc = 0; kc < DIM / 32; ++kc) {
            // A 16x32 bf16 layout: elems0-15 -> K=h*16+0..15 (contiguous, matches B)
            v8bf alo = *(const v8bf*)(arow + kc * 32 + h * 16);
            v8bf ahi = *(const v8bf*)(arow + kc * 32 + h * 16 + 8);
            v16bf a = __builtin_shufflevector(alo, ahi,
                        0,1,2,3,4,5,6,7,8,9,10,11,12,13,14,15);
            // B 32x16 bf16 layout: N=lane&15, K=h*16+e (one contiguous 32B run)
            v16bf bb = *(const v16bf*)(brow + kc * 32 + h * 16);
            acc = __builtin_amdgcn_wmma_f32_16x16x32_bf16(
                      false, a, false, bb, (short)0, acc, false, false);
        }

        // ---- fused epilogue: +bias, hw tanh, *v[e]; lane-private accumulation
        int   e  = et * 16 + m;              // column index (N) of C
        float be = biasb[e];
        float ve = v[e];
#pragma unroll
        for (int r = 0; r < 8; ++r)
            sAcc[r] += FAST_TANH(acc[r] + be) * ve;
    }

    // single cross-lane reduction at the end (sum within each 16-lane group)
#pragma unroll
    for (int r = 0; r < 8; ++r) {
        float t = sAcc[r];
        t += __shfl_xor(t, 1, 32);
        t += __shfl_xor(t, 2, 32);
        t += __shfl_xor(t, 4, 32);
        t += __shfl_xor(t, 8, 32);
        sAcc[r] = t;
    }

    // C layout: lanes 0-15 -> rows M=r ; lanes 16-31 -> rows M=8+r
    size_t base = (size_t)b * SRC + (size_t)st * 128 + wave * 16;
    if (lane == 0) {
#pragma unroll
        for (int r = 0; r < 8; ++r) scores[base + r] = sAcc[r];
    } else if (lane == 16) {
#pragma unroll
        for (int r = 0; r < 8; ++r) scores[base + 8 + r] = sAcc[r];
    }
}

// ---------------------------------------------------------------------------
// K2: in-place row softmax over SRC=2048 (one block per batch row)
// ---------------------------------------------------------------------------
__global__ __launch_bounds__(256)
void softmax_kernel(float* __restrict__ align) {
    __shared__ float red[256];
    int b = blockIdx.x, tid = threadIdx.x;
    float* row = align + (size_t)b * SRC;
    float vals[8];
    float mx = -INFINITY;
#pragma unroll
    for (int i = 0; i < 8; ++i) { vals[i] = row[tid + i * 256]; mx = fmaxf(mx, vals[i]); }
    red[tid] = mx; __syncthreads();
    for (int s = 128; s > 0; s >>= 1) {
        if (tid < s) red[tid] = fmaxf(red[tid], red[tid + s]);
        __syncthreads();
    }
    mx = red[0]; __syncthreads();
    float sum = 0.f;
#pragma unroll
    for (int i = 0; i < 8; ++i) { vals[i] = __expf(vals[i] - mx); sum += vals[i]; }
    red[tid] = sum; __syncthreads();
    for (int s = 128; s > 0; s >>= 1) {
        if (tid < s) red[tid] += red[tid + s];
        __syncthreads();
    }
    float inv = 1.0f / red[0];
#pragma unroll
    for (int i = 0; i < 8; ++i) row[tid + i * 256] = vals[i] * inv;
}

// ---------------------------------------------------------------------------
// K3: c[b,d] = sum_s p[b,s] * mb[b,s,d]   (streaming, float4-coalesced)
// ---------------------------------------------------------------------------
__global__ __launch_bounds__(256)
void context_kernel(const float* __restrict__ mb, const float* __restrict__ p,
                    float* __restrict__ c) {
    int b  = blockIdx.x;
    int d4 = threadIdx.x * 4;                 // 256 threads cover DIM=1024
    const float* prow = p  + (size_t)b * SRC;
    const float* mbb  = mb + (size_t)b * SRC * DIM;
    float ax = 0.f, ay = 0.f, az = 0.f, aw = 0.f;
#pragma unroll 4
    for (int s = 0; s < SRC; ++s) {
        float w = prow[s];                    // uniform -> scalar load
        float4 hv = *(const float4*)(mbb + (size_t)s * DIM + d4);
        ax += w * hv.x; ay += w * hv.y; az += w * hv.z; aw += w * hv.w;
    }
    float4 out; out.x = ax; out.y = ay; out.z = az; out.w = aw;
    *(float4*)(c + (size_t)b * DIM + d4) = out;
}

// ---------------------------------------------------------------------------
extern "C" void kernel_launch(void* const* d_in, const int* in_sizes, int n_in,
                              void* d_out, int out_size, void* d_ws, size_t ws_size,
                              hipStream_t stream) {
    const float* source    = (const float*)d_in[0];
    const float* memory    = (const float*)d_in[1];
    const float* condition = (const float*)d_in[2];
    const float* W_query   = (const float*)d_in[3];
    const float* b_query   = (const float*)d_in[4];
    const float* W_context = (const float*)d_in[5];
    const float* W_cond    = (const float*)d_in[6];
    const float* v         = (const float*)d_in[7];

    float* out_c     = (float*)d_out;                    // [64,1024]
    float* out_align = (float*)d_out + BATCH * DIM;      // [64,2048] (== [64,1,2048])
    float* bias      = (float*)d_ws;                     // [64,1024] scratch (256 KB)

    bias_kernel<<<BATCH * DIM / 256, 256, 0, stream>>>(
        source, condition, W_query, b_query, W_cond, bias);

    dim3 g1(BATCH, SRC / 128);
    score_kernel<<<g1, 256, 0, stream>>>(memory, W_context, bias, v, out_align);

    softmax_kernel<<<BATCH, 256, 0, stream>>>(out_align);

    context_kernel<<<BATCH, 256, 0, stream>>>(memory, out_align, out_c);
}
